// HierarchicalWaveLayer_19207093747861
// MI455X (gfx1250) — matrix operand order repaired and checked by reading, not compiled, not run on hardware
//
#include <hip/hip_runtime.h>
#include <hip/hip_bf16.h>
#include <math.h>

// ---------------------------------------------------------------------------
// HierarchicalWaveLayer forward for MI455X (gfx1250, wave32, WMMA + TDM).
//
//   k0: setup        -> lam_r/lam_i + 4x4 coupling matrix
//   k1: conv_x       -> x (f32) -> hi/lo bf16 planes [16384][1024]
//   k2: conv_w       -> [W_gate|W_b_real|W_b_imag] -> hi/lo bf16, transposed [n][d]
//   k3: fused GEMM   -> G[16384][1536], bf16x3 split WMMA, A/B tiles staged by
//                       tensor_load_to_lds (double buffered, TENSORcnt waits,
//                       hardware pad 64B->80B rows), sigmoid on gate cols
//   k4: chunk carries-> per-(b,chunk,k,h) zero-init scan over 64 steps
//   k5: carry combine-> sequential 32-chunk combine with lam^64
//   k6: output scan  -> re-scan with carries, gate*Bu, 4x4 coupling, output
// ---------------------------------------------------------------------------

typedef __bf16 bf16;
typedef __attribute__((ext_vector_type(16))) __bf16 v16bf;
typedef __attribute__((ext_vector_type(8)))  __bf16 v8bf;
typedef __attribute__((ext_vector_type(4)))  __bf16 v4bf;
typedef __attribute__((ext_vector_type(8)))  float   v8f;
typedef __attribute__((ext_vector_type(4)))  unsigned int u32x4;
typedef __attribute__((ext_vector_type(8)))  int i32x8;
typedef __attribute__((ext_vector_type(4)))  int i32x4;

#define BATCH   8
#define SEQ     2048
#define DM      1024
#define NSC     4
#define DST     128
#define ROWS    (BATCH*SEQ)      // 16384
#define NCOL    1536             // 512 gate | 512 Bu_r | 512 Bu_i
#define MT      128
#define NT      128
#define KT      32
#define NK      (DM/KT)          // 32
#define LDA     40               // padded LDS row stride in bf16 (80B)
#define NCHUNK  32
#define CLEN    64

// workspace layout (floats)
#define LAM_OFF   ((size_t)ROWS*NCOL)            // 25,165,824
#define CM_OFF    (LAM_OFF + 1024)
#define CARRY_OFF (CM_OFF + 32)
#define XH_OFF    (CARRY_OFF + (size_t)BATCH*NCHUNK*NSC*2*DST)   // +262,144
#define XL_OFF    (XH_OFF + (size_t)ROWS*DM/2)   // bf16 plane = ROWS*DM/2 floats
#define WH_OFF    (XL_OFF + (size_t)ROWS*DM/2)
#define WL_OFF    (WH_OFF + (size_t)NCOL*DM/2)
// total ~ 43.8M floats ~ 176 MB workspace

__device__ static inline v8f wmma_bf16(v16bf a, v16bf b, v8f c) {
  return __builtin_amdgcn_wmma_f32_16x16x32_bf16(false, a, false, b,
                                                 (short)0, c, false, false);
}

// TDM: 2-D tile load global->LDS, 128 rows x 64 bytes, global row stride 2048B,
// LDS padded +16B per 64B row (pad_interval=16 dwords, pad_amount=4 dwords).
__device__ static inline void tdm_tile_load(unsigned lds_addr,
                                            unsigned long long ga) {
  u32x4 g0;
  g0[0] = 1u;                                            // count=1 (user D#)
  g0[1] = lds_addr;                                      // LDS byte address
  g0[2] = (unsigned)ga;                                  // global_addr[31:0]
  g0[3] = (unsigned)((ga >> 32) & 0x01FFFFFFu) | (2u << 30); // [56:32] | type=2
  i32x8 g1;
  g1[0] = (1 << 20) | (3 << 22) | (3 << 25); // pad_en, interval=16dw, amount=4dw
  g1[1] = 0;                                 // abar=0, tensor_dim0 low16 (dim0=1<<16)
  g1[2] = 1;                                 // tensor_dim0 hi16=1, tensor_dim1 lo16=0
  g1[3] = 1 | (64 << 16);                    // tensor_dim1 hi16=1, tile_dim0=64 B
  g1[4] = 128;                               // tile_dim1=128 rows, tile_dim2=0
  g1[5] = 2048;                              // tensor_dim0_stride = 2048 B
  g1[6] = 0;
  g1[7] = 0;
  i32x4 z4 = {0, 0, 0, 0};                   // 2-D: groups 2/3 unused
  i32x8 z8 = {0, 0, 0, 0, 0, 0, 0, 0};
  __builtin_amdgcn_tensor_load_to_lds(g0, g1, z4, z4, z8, 0);
}

// ---------------------------------------------------------------- k0: setup
__global__ void wave_setup_kernel(const float* __restrict__ nu,
                                  const float* __restrict__ lth,
                                  const float* __restrict__ sigma,
                                  float* __restrict__ ws) {
  int t = threadIdx.x;
  if (t < 512) {
    float s  = 1.f / (1.f + __expf(-nu[t]));
    float r  = 0.9f + 0.099f * s;
    float th = __expf(lth[t]);
    ws[LAM_OFF + t]       = r * __cosf(th);
    ws[LAM_OFF + 512 + t] = r * __sinf(th);
  }
  if (t < 16) {
    float thv[4];
    thv[0] = 0.0063f;
    thv[1] = 0.063f;
    thv[2] = __expf(0.5f * (__logf(0.063f) + __logf(1.257f)));
    thv[3] = 1.257f;
    int i = t >> 2, j = t & 3;
    float inv = 1.f / (fabsf(sigma[0]) + 1e-6f);
    float s = 0.f;
    for (int q = 0; q < 4; q++) s += __expf(-fabsf(thv[i] - thv[q]) * inv);
    ws[CM_OFF + t] = __expf(-fabsf(thv[i] - thv[j]) * inv) / s;
  }
}

// ------------------------------------------- k1: x -> hi/lo bf16 planes
__global__ __launch_bounds__(256)
void wave_convx_kernel(const float* __restrict__ x, float* __restrict__ ws) {
  size_t i = (size_t)blockIdx.x * 256 + threadIdx.x;   // one float4 each
  float4 v = ((const float4*)x)[i];
  bf16* xh = (bf16*)(ws + XH_OFF);
  bf16* xl = (bf16*)(ws + XL_OFF);
  v4bf h, l;
  h[0] = (bf16)v.x; h[1] = (bf16)v.y; h[2] = (bf16)v.z; h[3] = (bf16)v.w;
  l[0] = (bf16)(v.x - (float)h[0]);
  l[1] = (bf16)(v.y - (float)h[1]);
  l[2] = (bf16)(v.z - (float)h[2]);
  l[3] = (bf16)(v.w - (float)h[3]);
  *(v4bf*)(xh + i*4) = h;
  *(v4bf*)(xl + i*4) = l;
}

// ------------------- k2: weights -> hi/lo bf16, transposed [n][d] (LDS tiles)
__global__ __launch_bounds__(256)
void wave_convw_kernel(const float* __restrict__ Wg,
                       const float* __restrict__ Wbr,
                       const float* __restrict__ Wbi,
                       float* __restrict__ ws) {
  __shared__ float tile[32][33];
  int n0 = blockIdx.x * 32;      // 0..1535
  int d0 = blockIdx.y * 32;      // 0..1023
  const float* src; size_t cbase; int cstride;
  if (n0 < 512)       { src = Wg;  cbase = (size_t)n0;            cstride = 512; }
  else if (n0 < 1024) { int k = (n0-512) >> 7;
                        src = Wbr; cbase = (size_t)k*DM*DST + ((n0-512)  & 127); cstride = DST; }
  else                { int k = (n0-1024) >> 7;
                        src = Wbi; cbase = (size_t)k*DM*DST + ((n0-1024) & 127); cstride = DST; }
  int tid = threadIdx.x;
#pragma unroll
  for (int rep = 0; rep < 4; rep++) {     // coalesced reads along n
    int idx = rep*256 + tid;
    int dd = idx >> 5, nn = idx & 31;
    tile[nn][dd] = src[cbase + (size_t)(d0 + dd)*cstride + nn];
  }
  __syncthreads();
  bf16* wh = (bf16*)(ws + WH_OFF);
  bf16* wl = (bf16*)(ws + WL_OFF);
#pragma unroll
  for (int rep = 0; rep < 4; rep++) {     // coalesced writes along d
    int idx = rep*256 + tid;
    int nn = idx >> 5, dd = idx & 31;
    float v = tile[nn][dd];
    bf16 hv = (bf16)v;
    size_t o = (size_t)(n0 + nn)*DM + d0 + dd;
    wh[o] = hv;
    wl[o] = (bf16)(v - (float)hv);
  }
}

// ------------------------------ k3: bf16x3 WMMA GEMM, TDM-staged, dbl-buffered
__global__ __launch_bounds__(256)
void wave_gemm_kernel(const float* __restrict__ bg, float* __restrict__ ws) {
  __shared__ __align__(16) bf16 sA[2][2][MT*LDA];   // [buf][hi/lo]
  __shared__ __align__(16) bf16 sB[2][2][NT*LDA];

  const int tid  = threadIdx.x;
  const int lane = tid & 31;
  const int wave = tid >> 5;
  const int half = lane >> 4;
  const int l16  = lane & 15;
  const int mw   = wave >> 1;      // 0..3
  const int nw   = wave & 1;       // 0..1

  const int nb = blockIdx.x;       // 0..11
  const int mb = blockIdx.y;       // 0..127
  const int n0 = nb * NT;
  const int m0 = mb * MT;

  const bf16* xh = (const bf16*)(ws + XH_OFF);
  const bf16* xl = (const bf16*)(ws + XL_OFF);
  const bf16* wh = (const bf16*)(ws + WH_OFF);
  const bf16* wl = (const bf16*)(ws + WL_OFF);

  // global tile bases (row stride 2048B; K advances 64B per iteration)
  unsigned long long gAh = (unsigned long long)(uintptr_t)(xh + (size_t)m0*DM);
  unsigned long long gAl = (unsigned long long)(uintptr_t)(xl + (size_t)m0*DM);
  unsigned long long gBh = (unsigned long long)(uintptr_t)(wh + (size_t)n0*DM);
  unsigned long long gBl = (unsigned long long)(uintptr_t)(wl + (size_t)n0*DM);
  // LDS byte addresses (low 32 bits of generic pointer = LDS offset)
  unsigned ldsA[2][2], ldsB[2][2];
#pragma unroll
  for (int b = 0; b < 2; b++)
#pragma unroll
    for (int p = 0; p < 2; p++) {
      ldsA[b][p] = (unsigned)(uintptr_t)(void*)&sA[b][p][0];
      ldsB[b][p] = (unsigned)(uintptr_t)(void*)&sB[b][p][0];
    }

  v8f zero = {};
  v8f acc[8];
#pragma unroll
  for (int i = 0; i < 8; i++) acc[i] = zero;

  if (wave == 0) {                       // prologue: tile 0 -> buffer 0
    tdm_tile_load(ldsA[0][0], gAh);
    tdm_tile_load(ldsA[0][1], gAl);
    tdm_tile_load(ldsB[0][0], gBh);
    tdm_tile_load(ldsB[0][1], gBl);
  }

  for (int kt = 0; kt < NK; kt++) {
    const int cur = kt & 1;
    __syncthreads();                     // all waves done reading buf 1-cur
    if (wave == 0) {
      if (kt + 1 < NK) {                 // stage next tile under this compute
        unsigned long long off = (unsigned long long)(kt + 1) * 64ull;
        tdm_tile_load(ldsA[1-cur][0], gAh + off);
        tdm_tile_load(ldsA[1-cur][1], gAl + off);
        tdm_tile_load(ldsB[1-cur][0], gBh + off);
        tdm_tile_load(ldsB[1-cur][1], gBl + off);
        __builtin_amdgcn_s_wait_tensorcnt((short)4);  // tile kt complete
      } else {
        __builtin_amdgcn_s_wait_tensorcnt((short)0);
      }
    }
    __syncthreads();                     // tile kt visible to all waves

    const bf16* pAh = &sA[cur][0][0];
    const bf16* pAl = &sA[cur][1][0];
    const bf16* pBh = &sB[cur][0][0];
    const bf16* pBl = &sB[cur][1][0];

    // A fragments: lane = row M=l16; K runs [8*half,+8) and [16+8*half,+8)
    v16bf Ah[2], Al[2];
#pragma unroll
    for (int im = 0; im < 2; im++) {
      int row  = mw*32 + im*16 + l16;
      int base = row*LDA + half*8;
      v8bf a0h = *(const v8bf*)&pAh[base];
      v8bf a1h = *(const v8bf*)&pAh[base + 16];
      v8bf a0l = *(const v8bf*)&pAl[base];
      v8bf a1l = *(const v8bf*)&pAl[base + 16];
#pragma unroll
      for (int e = 0; e < 8; e++) {
        Ah[im][e] = a0h[e]; Ah[im][8+e] = a1h[e];
        Al[im][e] = a0l[e]; Al[im][8+e] = a1l[e];
      }
    }
    // B fragments: lane = col N=l16; K runs [8*half,+8) and [16+8*half,+8) in [n][k]
#pragma unroll
    for (int jn = 0; jn < 4; jn++) {
      int n  = nw*64 + jn*16 + l16;
      int bb = n*LDA + half*8;
      v8bf b0h = *(const v8bf*)&pBh[bb];
      v8bf b1h = *(const v8bf*)&pBh[bb + 16];
      v8bf b0l = *(const v8bf*)&pBl[bb];
      v8bf b1l = *(const v8bf*)&pBl[bb + 16];
      v16bf Bh, Bl;
#pragma unroll
      for (int e = 0; e < 8; e++) {
        Bh[e] = b0h[e]; Bh[8+e] = b1h[e];
        Bl[e] = b0l[e]; Bl[8+e] = b1l[e];
      }
#pragma unroll
      for (int im = 0; im < 2; im++) {
        int ci = im*4 + jn;
        acc[ci] = wmma_bf16(Ah[im], Bh, acc[ci]);   // hi*hi
        acc[ci] = wmma_bf16(Ah[im], Bl, acc[ci]);   // hi*lo
        acc[ci] = wmma_bf16(Al[im], Bh, acc[ci]);   // lo*hi
      }
    }
  }

  // epilogue: C/D layout -> VGPR r: row = base + r + 8*half, col = l16
  float* G = ws;
#pragma unroll
  for (int im = 0; im < 2; im++) {
#pragma unroll
    for (int jn = 0; jn < 4; jn++) {
      v8f c = acc[im*4 + jn];
      int row0 = m0 + mw*32 + im*16 + half*8;
      int col  = n0 + nw*64 + jn*16 + l16;
      bool isGate = (col < 512);
      float bias = isGate ? bg[col] : 0.f;
#pragma unroll
      for (int r = 0; r < 8; r++) {
        float v = c[r];
        if (isGate) v = 1.f / (1.f + __expf(-(v + bias)));
        G[(size_t)(row0 + r)*NCOL + col] = v;
      }
    }
  }
}

// ----------------------------------------------- k4: per-chunk zero-init scan
__global__ __launch_bounds__(128)
void wave_carry_kernel(float* __restrict__ ws) {
  int h  = threadIdx.x;
  int bc = blockIdx.x;
  int b  = bc >> 5, ch = bc & 31;
  float lr[NSC], li[NSC], hr[NSC], hi_[NSC];
#pragma unroll
  for (int k = 0; k < NSC; k++) {
    lr[k] = ws[LAM_OFF + k*DST + h];
    li[k] = ws[LAM_OFF + 512 + k*DST + h];
    hr[k] = 0.f; hi_[k] = 0.f;
  }
  const float* Grow = ws + (size_t)(b*SEQ + ch*CLEN)*NCOL;
  for (int t = 0; t < CLEN; t++) {
#pragma unroll
    for (int k = 0; k < NSC; k++) {
      float g  = Grow[k*DST + h];
      float ur = g * Grow[512  + k*DST + h];
      float ui = g * Grow[1024 + k*DST + h];
      float nr = lr[k]*hr[k] - li[k]*hi_[k] + ur;
      float ni = lr[k]*hi_[k] + li[k]*hr[k] + ui;
      hr[k] = nr; hi_[k] = ni;
    }
    Grow += NCOL;
  }
#pragma unroll
  for (int k = 0; k < NSC; k++) {
    size_t o = CARRY_OFF + ((size_t)(bc*NSC + k)*2)*DST + h;
    ws[o]       = hr[k];
    ws[o + DST] = hi_[k];
  }
}

// ----------------------------- k5: sequential carry combine (lam^CLEN factors)
__global__ __launch_bounds__(512)
void wave_combine_kernel(float* __restrict__ ws) {
  int b = blockIdx.x;
  int k = threadIdx.x >> 7;
  int h = threadIdx.x & 127;
  float Pr = ws[LAM_OFF + k*DST + h];
  float Pi = ws[LAM_OFF + 512 + k*DST + h];
#pragma unroll
  for (int i = 0; i < 6; i++) {
    float nr = Pr*Pr - Pi*Pi, ni = 2.f*Pr*Pi;
    Pr = nr; Pi = ni;
  }
  float cr = 0.f, ci = 0.f;
  for (int j = 0; j < NCHUNK; j++) {
    size_t o = CARRY_OFF + ((size_t)((b*NCHUNK + j)*NSC + k)*2)*DST + h;
    float ar = ws[o], ai = ws[o + DST];
    ws[o] = cr; ws[o + DST] = ci;
    float nr = Pr*cr - Pi*ci + ar;
    float ni = Pr*ci + Pi*cr + ai;
    cr = nr; ci = ni;
  }
}

// ----------------------- k6: re-scan with carries, gate*Bu, coupling, output
__global__ __launch_bounds__(128)
void wave_output_kernel(const float* __restrict__ ws, float* __restrict__ out) {
  int h  = threadIdx.x;
  int bc = blockIdx.x;
  int b  = bc >> 5, ch = bc & 31;
  float lr[NSC], li[NSC], hr[NSC], hi_[NSC], cm[16];
#pragma unroll
  for (int k = 0; k < NSC; k++) {
    lr[k] = ws[LAM_OFF + k*DST + h];
    li[k] = ws[LAM_OFF + 512 + k*DST + h];
    size_t o = CARRY_OFF + ((size_t)(bc*NSC + k)*2)*DST + h;
    hr[k] = ws[o]; hi_[k] = ws[o + DST];
  }
#pragma unroll
  for (int i = 0; i < 16; i++) cm[i] = ws[CM_OFF + i];

  const float* Grow = ws  + (size_t)(b*SEQ + ch*CLEN)*NCOL;
  float*       Orow = out + (size_t)(b*SEQ + ch*CLEN)*(NSC*2*DST);
  for (int t = 0; t < CLEN; t++) {
#pragma unroll
    for (int k = 0; k < NSC; k++) {
      float g  = Grow[k*DST + h];
      float ur = g * Grow[512  + k*DST + h];
      float ui = g * Grow[1024 + k*DST + h];
      float nr = lr[k]*hr[k] - li[k]*hi_[k] + ur;
      float ni = lr[k]*hi_[k] + li[k]*hr[k] + ui;
      hr[k] = nr; hi_[k] = ni;
    }
#pragma unroll
    for (int i = 0; i < NSC; i++) {
      float orr = 0.f, oii = 0.f;
#pragma unroll
      for (int j = 0; j < NSC; j++) { orr += cm[i*4+j]*hr[j]; oii += cm[i*4+j]*hi_[j]; }
      Orow[i*256 + h]       = orr;
      Orow[i*256 + 128 + h] = oii;
    }
    Grow += NCOL;
    Orow += NSC*2*DST;
  }
}

// ---------------------------------------------------------------------------
extern "C" void kernel_launch(void* const* d_in, const int* in_sizes, int n_in,
                              void* d_out, int out_size, void* d_ws, size_t ws_size,
                              hipStream_t stream) {
  (void)in_sizes; (void)n_in; (void)out_size; (void)ws_size;
  const float* x   = (const float*)d_in[0];
  const float* Wbr = (const float*)d_in[1];
  const float* Wbi = (const float*)d_in[2];
  const float* Wg  = (const float*)d_in[3];
  const float* bg  = (const float*)d_in[4];
  const float* nu  = (const float*)d_in[5];
  const float* lth = (const float*)d_in[6];
  const float* sg  = (const float*)d_in[7];
  float* ws  = (float*)d_ws;
  float* out = (float*)d_out;

  wave_setup_kernel<<<1, 512, 0, stream>>>(nu, lth, sg, ws);
  wave_convx_kernel<<<ROWS*DM/1024, 256, 0, stream>>>(x, ws);
  wave_convw_kernel<<<dim3(NCOL/32, DM/32), 256, 0, stream>>>(Wg, Wbr, Wbi, ws);
  wave_gemm_kernel<<<dim3(NCOL/NT, ROWS/MT), 256, 0, stream>>>(bg, ws);
  wave_carry_kernel<<<BATCH*NCHUNK, 128, 0, stream>>>(ws);
  wave_combine_kernel<<<BATCH, 512, 0, stream>>>(ws);
  wave_output_kernel<<<BATCH*NCHUNK, 128, 0, stream>>>(ws, out);
}
